// DiscriminativeLoss_65403761984267
// MI455X (gfx1250) — matrix operand order repaired and checked by reading, not compiled, not run on hardware
//
#include <hip/hip_runtime.h>
#include <hip/hip_bf16.h>

typedef __attribute__((ext_vector_type(16))) _Float16 v16h;
typedef __attribute__((ext_vector_type(8)))  _Float16 v8h;
typedef __attribute__((ext_vector_type(8)))  float    v8f;

#define NUM_CLASSES 33
#define NFEAT 5

// workspace float layout
#define ACC_OFF   0      // acc[c*6 + n] : n<5 feature sums, n==5 count   (33*6 = 198)
#define MEANS_OFF 198    // means[c*5 + f]                                (165)
#define VARS_OFF  363    // varsums[c]                                    (33)
#define STAT_OFF  396    // stat[0]=beta*dist+gamma*reg, stat[1]=n_unique
#define WS_FLOATS 400

// One-hot A staging: 48 real class rows (3 tiles x 16) + row 48 = dummy sink.
#define AROWS   50                 // padded
#define ASTRIDE (AROWS * 32)       // halves per wave region (1600 -> 3200 B)

// compiler-only fence: hardware keeps same-wave LDS ops in order (DScnt, in-order)
#define WAVE_FENCE() asm volatile("" ::: "memory")

__global__ void zero_ws_kernel(float* ws) {
    int i = threadIdx.x;
    if (i < WS_FLOATS) ws[i] = 0.0f;
}

// A tile (16x32 f16): lane row = lane%16; halves 0-7 = K k0a..k0a+7,
// halves 8-15 = K k0a+16..k0a+23  (k0a = 0 for lanes<16, 8 for lanes>=16)
__device__ __forceinline__ v16h load_a_tile(const _Float16* A, int row, int k0a) {
    const v8h lo = *(const v8h*)(A + row * 32 + k0a);
    const v8h hi = *(const v8h*)(A + row * 32 + k0a + 16);
    return __builtin_shufflevector(lo, hi, 0,1,2,3,4,5,6,7,8,9,10,11,12,13,14,15);
}

// B tile (32x16 f16): lane col = lane%16; halves 0-7 = K k0b..k0b+7, halves 8-15 = K k0b+16..k0b+23
__device__ __forceinline__ v16h load_b_tile(const _Float16* Bm, int col, int k0b) {
    const _Float16* p = Bm + col * 32 + k0b;
    const v8h lo = *(const v8h*)(p);
    const v8h hi = *(const v8h*)(p + 16);
    return __builtin_shufflevector(lo, hi, 0,1,2,3,4,5,6,7,8,9,10,11,12,13,14,15);
}

// ---------------------------------------------------------------------------
// Pass 1: segment sums via one-hot WMMA (one-hot built by LDS scatter).
// D[m,n] = sum_k onehot(label[k]==classBase+m) * B[k,n]
// B columns: n=0..4 features, n=5 ones (counts), n>=6 zero (pre-zeroed).
// ---------------------------------------------------------------------------
__global__ __launch_bounds__(256)
void seg_sums_wmma(const float* __restrict__ x, const long long* __restrict__ tgt,
                   float* __restrict__ acc, int N) {
    __shared__ __align__(16) _Float16 sA[8][ASTRIDE];
    __shared__ __align__(16) _Float16 sB[8][16 * 32];   // full 16 columns, 6 used

    const int lane = threadIdx.x & 31;
    const int wid  = threadIdx.x >> 5;
    _Float16* A  = &sA[wid][0];
    _Float16* Bm = &sB[wid][0];

    // zero this wave's A + B regions once (dword stores; same-wave in-order)
    {
        unsigned* Az = (unsigned*)A;
        #pragma unroll
        for (int i = 0; i < ASTRIDE / 2; i += 32) Az[i + lane] = 0u;
        unsigned* Bz = (unsigned*)Bm;
        #pragma unroll
        for (int i = 0; i < (16 * 32) / 2; i += 32) Bz[i + lane] = 0u;
    }
    WAVE_FENCE();

    const int gwave   = blockIdx.x * 8 + wid;
    const int nwaves  = gridDim.x * 8;
    const int nchunks = (N + 31) >> 5;

    const int r16  = lane & 15;
    const int ncol = lane & 15;
    const int k0a  = (lane & 16) ? 8 : 0;
    const int k0b  = (lane & 16) ? 8 : 0;

    // per-feature stream bases (SGPR bases -> GVS scale_offset addressing)
    const float* x0 = x;
    const float* x1 = x + 1 * (size_t)N;
    const float* x2 = x + 2 * (size_t)N;
    const float* x3 = x + 3 * (size_t)N;
    const float* x4 = x + 4 * (size_t)N;

    v8f c0 = {}, c1 = {}, c2 = {};
    int prev_row = 48;

    for (int chunk = gwave; chunk < nchunks; chunk += nwaves) {
        const int idx = chunk * 32 + lane;
        const bool valid = idx < N;
        const unsigned ci = valid ? (unsigned)idx : (unsigned)(N - 1);

        const int lab = (int)tgt[ci];           // b64 load, low dword used
        const float f0 = x0[ci], f1 = x1[ci], f2 = x2[ci], f3 = x3[ci], f4 = x4[ci];

        int row = (unsigned)lab > 47u ? 48 : lab;
        row = valid ? row : 48;                 // dummy sink row

        // one-hot scatter: clear last position, set new one (same-wave, in-order)
        A[prev_row * 32 + lane] = (_Float16)0;
        A[row * 32 + lane]      = (_Float16)1;
        prev_row = row;

        // B staging, feature-major
        Bm[0*32 + lane] = (_Float16)f0;
        Bm[1*32 + lane] = (_Float16)f1;
        Bm[2*32 + lane] = (_Float16)f2;
        Bm[3*32 + lane] = (_Float16)f3;
        Bm[4*32 + lane] = (_Float16)f4;
        Bm[5*32 + lane] = valid ? (_Float16)1 : (_Float16)0;
        WAVE_FENCE();

        const v16h b  = load_b_tile(Bm, ncol, k0b);   // unconditional: cols 6..15 are zero
        const v16h a0 = load_a_tile(A, r16,      k0a);
        const v16h a1 = load_a_tile(A, 16 + r16, k0a);
        const v16h a2 = load_a_tile(A, 32 + r16, k0a);

        c0 = __builtin_amdgcn_wmma_f32_16x16x32_f16(false, a0, false, b, (short)0, c0, false, false);
        c1 = __builtin_amdgcn_wmma_f32_16x16x32_f16(false, a1, false, b, (short)0, c1, false, false);
        c2 = __builtin_amdgcn_wmma_f32_16x16x32_f16(false, a2, false, b, (short)0, c2, false, false);
        WAVE_FENCE();
    }

    // C layout: VGPR r -> M = r + (lane>=16 ? 8 : 0), N = lane%16
    if (ncol < 6) {
        const int mbase = (lane & 16) ? 8 : 0;
        #pragma unroll
        for (int r = 0; r < 8; ++r) {
            const int mm = mbase + r;
            atomicAdd(&acc[(mm)      * 6 + ncol], c0[r]);
            atomicAdd(&acc[(16 + mm) * 6 + ncol], c1[r]);
            if (mm == 0) atomicAdd(&acc[32 * 6 + ncol], c2[r]);
        }
    }
}

// ---------------------------------------------------------------------------
// Pass 2: tiny cluster-level math (single thread; nanoseconds of work).
// ---------------------------------------------------------------------------
__global__ void cluster_stats(const float* __restrict__ acc,
                              float* __restrict__ means, float* __restrict__ stat) {
    if (threadIdx.x != 0 || blockIdx.x != 0) return;
    float ms[NUM_CLASSES][NFEAT];
    bool  nz[NUM_CLASSES];
    float n_unique = 0.f, n_c = 0.f;
    for (int c = 0; c < NUM_CLASSES; ++c) {
        const float cnt  = acc[c * 6 + 5];
        const float safe = fmaxf(cnt, 1.0f);
        const bool present = cnt > 0.0f;
        if (present) n_unique += 1.0f;
        nz[c] = present && (c != 0);
        if (nz[c]) n_c += 1.0f;
        for (int f = 0; f < NFEAT; ++f) {
            const float mu = acc[c * 6 + f] / safe;
            means[c * NFEAT + f] = mu;
            ms[c][f] = nz[c] ? mu : 0.0f;
        }
    }
    float dist = 0.f;
    for (int i = 0; i < NUM_CLASSES; ++i) {
        if (!nz[i]) continue;
        for (int j = 0; j < NUM_CLASSES; ++j) {
            if (i == j || !nz[j]) continue;
            float d = 0.f;
            for (int f = 0; f < NFEAT; ++f) d += fabsf(ms[i][f] - ms[j][f]);
            const float h = fmaxf(2.0f * 1.5f - d, 0.0f);
            dist += h * h;
        }
    }
    const float dist_term = (n_c > 1.5f) ? dist / (n_c * (n_c - 1.0f)) : 0.0f;
    float reg = 0.f;
    for (int c = 0; c < NUM_CLASSES; ++c) {
        if (!nz[c]) continue;
        for (int f = 0; f < NFEAT; ++f) reg += fabsf(ms[c][f]);
    }
    const float reg_term = (n_c > 0.f) ? reg / n_c / n_c : 0.0f;
    stat[0] = 1.0f * dist_term + 0.001f * reg_term;   // BETA, GAMMA
    stat[1] = (n_unique > 0.f) ? n_unique : 1.0f;
}

// ---------------------------------------------------------------------------
// Pass 3: per-point variance, segment-summed per class via one-hot WMMA
// (single nonzero B column n==0 holding var values; cols 1..15 pre-zeroed).
// ---------------------------------------------------------------------------
__global__ __launch_bounds__(256)
void var_wmma(const float* __restrict__ x, const long long* __restrict__ tgt,
              const float* __restrict__ means, float* __restrict__ varsums, int N) {
    __shared__ __align__(16) _Float16 sA[8][ASTRIDE];
    __shared__ __align__(16) _Float16 sV[8][16 * 32];  // full 16 columns, col 0 used
    __shared__ float s_means[(NUM_CLASSES + 1) * NFEAT];  // row 0 zeroed, row 33 = zero sink

    // rows 1..32 from global; rows 0 and 33 zero (background / invalid -> cm = 0)
    for (int i = threadIdx.x; i < (NUM_CLASSES + 1) * NFEAT; i += blockDim.x)
        s_means[i] = (i >= NFEAT && i < NUM_CLASSES * NFEAT) ? means[i] : 0.0f;
    __syncthreads();   // one block barrier; everything after is intra-wave

    const int lane = threadIdx.x & 31;
    const int wid  = threadIdx.x >> 5;
    _Float16* A = &sA[wid][0];
    _Float16* V = &sV[wid][0];

    {
        unsigned* Az = (unsigned*)A;
        #pragma unroll
        for (int i = 0; i < ASTRIDE / 2; i += 32) Az[i + lane] = 0u;
        unsigned* Vz = (unsigned*)V;
        #pragma unroll
        for (int i = 0; i < (16 * 32) / 2; i += 32) Vz[i + lane] = 0u;
    }
    WAVE_FENCE();

    const int gwave   = blockIdx.x * 8 + wid;
    const int nwaves  = gridDim.x * 8;
    const int nchunks = (N + 31) >> 5;

    const int r16  = lane & 15;
    const int ncol = lane & 15;
    const int k0a  = (lane & 16) ? 8 : 0;
    const int k0b  = (lane & 16) ? 8 : 0;

    const float* x0 = x;
    const float* x1 = x + 1 * (size_t)N;
    const float* x2 = x + 2 * (size_t)N;
    const float* x3 = x + 3 * (size_t)N;
    const float* x4 = x + 4 * (size_t)N;

    v8f c0 = {}, c1 = {}, c2 = {};
    int prev_row = 48;

    for (int chunk = gwave; chunk < nchunks; chunk += nwaves) {
        const int idx = chunk * 32 + lane;
        const bool valid = idx < N;
        const unsigned ci = valid ? (unsigned)idx : (unsigned)(N - 1);

        const int lab = (int)tgt[ci];
        const unsigned mlab = ((unsigned)lab > 33u) ? 33u : (unsigned)lab;  // rows 0/33 are zero
        const float* cm = &s_means[mlab * NFEAT];

        float s = 0.f;
        s += fabsf(x0[ci] - cm[0]);
        s += fabsf(x1[ci] - cm[1]);
        s += fabsf(x2[ci] - cm[2]);
        s += fabsf(x3[ci] - cm[3]);
        s += fabsf(x4[ci] - cm[4]);
        const float h = fmaxf(s - 0.5f, 0.0f);   // DELTA_VAR
        float v = h * h;
        v = valid ? v : 0.0f;

        int row = (unsigned)lab > 47u ? 48 : lab;
        row = valid ? row : 48;

        A[prev_row * 32 + lane] = (_Float16)0;
        A[row * 32 + lane]      = (_Float16)1;
        prev_row = row;
        V[lane] = (_Float16)v;                   // column 0
        WAVE_FENCE();

        const v16h b  = load_b_tile(V, ncol, k0b);    // cols 1..15 are zero
        const v16h a0 = load_a_tile(A, r16,      k0a);
        const v16h a1 = load_a_tile(A, 16 + r16, k0a);
        const v16h a2 = load_a_tile(A, 32 + r16, k0a);

        c0 = __builtin_amdgcn_wmma_f32_16x16x32_f16(false, a0, false, b, (short)0, c0, false, false);
        c1 = __builtin_amdgcn_wmma_f32_16x16x32_f16(false, a1, false, b, (short)0, c1, false, false);
        c2 = __builtin_amdgcn_wmma_f32_16x16x32_f16(false, a2, false, b, (short)0, c2, false, false);
        WAVE_FENCE();
    }

    if (ncol == 0) {  // only column 0 carries data
        const int mbase = (lane & 16) ? 8 : 0;
        #pragma unroll
        for (int r = 0; r < 8; ++r) {
            const int mm = mbase + r;
            atomicAdd(&varsums[mm],      c0[r]);
            atomicAdd(&varsums[16 + mm], c1[r]);
            if (mm == 0) atomicAdd(&varsums[32], c2[r]);
        }
    }
}

__global__ void finalize_kernel(const float* __restrict__ acc,
                                const float* __restrict__ varsums,
                                const float* __restrict__ stat,
                                float* __restrict__ out) {
    if (threadIdx.x != 0 || blockIdx.x != 0) return;
    float vt = 0.f;
    for (int c = 1; c < NUM_CLASSES; ++c) {
        const float cnt = acc[c * 6 + 5];
        if (cnt > 0.0f) vt += varsums[c] / fmaxf(cnt, 1.0f);
    }
    vt /= stat[1];
    out[0] = 1.0f * vt + stat[0];   // ALPHA * var_term + (beta*dist + gamma*reg)
}

extern "C" void kernel_launch(void* const* d_in, const int* in_sizes, int n_in,
                              void* d_out, int out_size, void* d_ws, size_t ws_size,
                              hipStream_t stream) {
    const float*     x   = (const float*)d_in[0];
    const long long* tgt = (const long long*)d_in[1];
    const int N = in_sizes[1];

    float* ws    = (float*)d_ws;
    float* acc   = ws + ACC_OFF;
    float* means = ws + MEANS_OFF;
    float* vars  = ws + VARS_OFF;
    float* stat  = ws + STAT_OFF;

    zero_ws_kernel<<<1, 512, 0, stream>>>(ws);
    seg_sums_wmma<<<512, 256, 0, stream>>>(x, tgt, acc, N);
    cluster_stats<<<1, 1, 0, stream>>>(acc, means, stat);
    var_wmma<<<512, 256, 0, stream>>>(x, tgt, means, vars, N);
    finalize_kernel<<<1, 1, 0, stream>>>(acc, vars, stat, (float*)d_out);
}
